// TimeframeEncoder_11355893531350
// MI455X (gfx1250) — compile-verified, matrix operand order checked by reading, not run on hardware
//
#include <hip/hip_runtime.h>

#define SEQLEN   512
#define INDIM    64
#define HID      256
#define LN_EPS   1e-5f

#define SH0 264   // LDS row stride (halves) for 256-wide activation buffers
#define SH1 520   // LDS row stride (halves) for 512-wide [h1|h0] buffer
#define XSTR 68   // LDS row stride (floats) for staged x tile (bank-conflict pad)
#define W0KT 10   // k-tiles for layer0 (K=320)
#define W1KT 16   // k-tiles for layer1 (K=512)

#if defined(__has_builtin)
#  if __has_builtin(__builtin_amdgcn_global_load_async_to_lds_b128)
#    define HAVE_ASYNC_LDS 1
#  endif
#endif
#ifndef HAVE_ASYNC_LDS
#  define HAVE_ASYNC_LDS 0
#endif

#if HAVE_ASYNC_LDS && __has_builtin(__builtin_amdgcn_s_wait_asynccnt)
#  define WAIT_ASYNC() __builtin_amdgcn_s_wait_asynccnt(0)
#elif HAVE_ASYNC_LDS
#  define WAIT_ASYNC() asm volatile("s_wait_asynccnt 0" ::: "memory")
#else
#  define WAIT_ASYNC() do {} while (0)
#endif

typedef __attribute__((ext_vector_type(16))) __bf16          v16bf;
typedef __attribute__((ext_vector_type(16))) unsigned short  v16u;
typedef __attribute__((ext_vector_type(8)))  float           v8f;
typedef __attribute__((ext_vector_type(4)))  int             v4i_t;
typedef __attribute__((address_space(1))) v4i_t*             gptr_v4i;
typedef __attribute__((address_space(3))) v4i_t*             lptr_v4i;

union AFrag {
  uint4 q[2];
  v16u  u;
  v16bf v;
};

__device__ __forceinline__ unsigned short f2bf(float f) {
  unsigned u = __float_as_uint(f);
  u += 0x7FFFu + ((u >> 16) & 1u);   // round-to-nearest-even
  return (unsigned short)(u >> 16);
}
__device__ __forceinline__ float bf2f(unsigned short h) {
  return __uint_as_float(((unsigned)h) << 16);
}
__device__ __forceinline__ float sanitize(float v) {
  if (__builtin_isnan(v)) return 0.f;
  if (__builtin_isinf(v)) return v > 0.f ? 10.f : -10.f;
  return v;
}
__device__ __forceinline__ float sigmoid_f(float x) {
  return 1.f / (1.f + __expf(-x));
}
__device__ __forceinline__ float tanh_f(float x) {
  return 1.f - 2.f / (__expf(2.f * x) + 1.f);
}
__device__ __forceinline__ v8f splat8(float b) {
  v8f r;
#pragma unroll
  for (int j = 0; j < 8; ++j) r[j] = b;
  return r;
}
__device__ __forceinline__ v8f wmma_bf16(v16bf a, v16bf b, v8f c) {
  return __builtin_amdgcn_wmma_f32_16x16x32_bf16(false, a, false, b, (short)0, c,
                                                 false, false);
}

// A fragment (16x32 bf16) from an LDS row-major activation buffer.
__device__ __forceinline__ v16bf load_afrag_lds(const unsigned short* base,
                                                int strideH, int row, int k0) {
  AFrag a;
  const unsigned short* p = base + row * strideH + k0;
  a.q[0] = *(const uint4*)(p);
  a.q[1] = *(const uint4*)(p + 16);
  return a.v;
}

// B fragment (32x16 bf16) from pre-packed weights: 32 lanes x 16 halves, contiguous.
__device__ __forceinline__ v16bf load_bfrag(const unsigned short* __restrict__ wp,
                                            int tile, int lane) {
  AFrag b;
  const uint4* p = (const uint4*)(wp + ((size_t)tile * 32 + lane) * 16);
  b.q[0] = p[0];
  b.q[1] = p[1];
  return b.v;
}

// A fragment whose K-range lies in the staged x tile (fp32 in LDS -> sanitize -> bf16)
__device__ __forceinline__ v16bf xfrag_from_lds(const float* xs, int row, int k0) {
  union { v16u u; v16bf v; } cv;
  const float* p = xs + row * XSTR + k0;
#pragma unroll
  for (int j = 0; j < 8; ++j) cv.u[j] = f2bf(sanitize(p[j]));
#pragma unroll
  for (int j = 0; j < 8; ++j) cv.u[8 + j] = f2bf(sanitize(p[16 + j]));
  return cv.v;
}

// Stage one timestep's 16x64 fp32 x tile into LDS (async DMA when available).
// One 16-byte chunk per thread (256 threads = 4KB tile).
__device__ __forceinline__ void stage_x(const float* __restrict__ x, int g, int t,
                                        float* xs, int tid) {
  int row = tid >> 4;        // 0..15
  int seg = tid & 15;        // 16B segment within the row
  const float* src = x + ((size_t)(g * 16 + row) * SEQLEN + t) * INDIM + seg * 4;
  float* dst = xs + row * XSTR + seg * 4;
#if HAVE_ASYNC_LDS
  __builtin_amdgcn_global_load_async_to_lds_b128((gptr_v4i)src, (lptr_v4i)dst, 0, 0);
#else
  *(float4*)dst = *(const float4*)src;
#endif
}

// ---------------- weight pre-pack: W(K x fanout) fp32 -> bf16 B-fragment tiles ------------
__global__ void pack_weights(const float* __restrict__ W, int fanout, int Kt, int Nt,
                             unsigned short* __restrict__ dst) {
  int tid  = blockIdx.x * blockDim.x + threadIdx.x;
  int lane = tid & 31;
  int tile = tid >> 5;
  if (tile >= Kt * Nt) return;
  int nt = tile / Kt, kt = tile % Kt;
  int n   = nt * 16 + (lane & 15);
  int khi = (lane >= 16) ? 8 : 0;
  unsigned short* out = dst + ((size_t)tile * 32 + lane) * 16;
#pragma unroll
  for (int j = 0; j < 16; ++j) {
    int p = j & 1, v = j >> 1;
    int base = (v < 4) ? 2 * v : 16 + 2 * (v - 4);
    int k = kt * 32 + base + p + khi;
    out[j] = f2bf(W[(size_t)k * fanout + n]);
  }
}

// ---------------- persistent GRU kernel: 1 WG = 16 batch rows, all 512 steps -------------
__global__ __launch_bounds__(256, 1)
void gru_persistent(const float* __restrict__ x,
                    const float* __restrict__ bz0, const float* __restrict__ bc0,
                    const float* __restrict__ bz1, const float* __restrict__ bc1,
                    const unsigned short* __restrict__ wz0p,
                    const unsigned short* __restrict__ wc0p,
                    const unsigned short* __restrict__ wz1p,
                    const unsigned short* __restrict__ wc1p,
                    const float* __restrict__ gamma, const float* __restrict__ beta,
                    float* __restrict__ out) {
  __shared__ __align__(16) unsigned short comb0h[16 * SH0];  // h0 (bf16)
  __shared__ __align__(16) unsigned short comb1[16 * SH1];   // [h1 | h0_new] (bf16)
  __shared__ __align__(16) unsigned short rc[16 * SH0];      // r*h (bf16)
  __shared__ float zbuf[16 * HID];                           // z gate (fp32)
  __shared__ __align__(16) float xstage[2][16 * XSTR];       // double-buffered x tile

  const int g    = blockIdx.x;        // batch-row group (16 rows)
  const int tid  = threadIdx.x;
  const int w    = tid >> 5;          // wave 0..7
  const int lane = tid & 31;
  const int mrow = lane & 15;         // tile row / output row base
  const int hi8  = ((lane >> 4) & 1) * 8;

  for (int i = tid; i < 16 * SH0; i += 256) { comb0h[i] = 0; rc[i] = 0; }
  for (int i = tid; i < 16 * SH1; i += 256) comb1[i] = 0;
  stage_x(x, g, 0, xstage[0], tid);        // prefetch first x tile
  WAIT_ASYNC();
  __syncthreads();

  // per-lane bias values for owned output columns (folded into accumulator init)
  float bzv0[4], bzv1[4], bcv0[2], bcv1[2];
#pragma unroll
  for (int nt = 0; nt < 4; ++nt) {
    int c = w * 64 + nt * 16 + mrow;       // gate col in 0..511 (z: 0..255, r: 256..511)
    bzv0[nt] = bz0[c];
    bzv1[nt] = bz1[c];
  }
#pragma unroll
  for (int nt = 0; nt < 2; ++nt) {
    int c = w * 32 + nt * 16 + mrow;       // candidate col in 0..255
    bcv0[nt] = bc0[c];
    bcv1[nt] = bc1[c];
  }

  for (int t = 0; t < SEQLEN; ++t) {
    const float* xs = xstage[t & 1];
    // x A-fragments (k-tiles 8,9 of layer-0 matmuls), reused by gates + candidate
    v16bf xf0 = xfrag_from_lds(xs, mrow, 0 + hi8);
    v16bf xf1 = xfrag_from_lds(xs, mrow, 32 + hi8);

    // async prefetch of next timestep's x tile (lands before end-of-loop barrier)
    {
      int tn = (t + 1 < SEQLEN) ? t + 1 : 0;
      stage_x(x, g, tn, xstage[(t + 1) & 1], tid);
    }

    // ---- layer0 gates: cols w*64..w*64+63 of [z|r] ----
    v8f gacc[4];
#pragma unroll
    for (int nt = 0; nt < 4; ++nt) gacc[nt] = splat8(bzv0[nt]);
    {
      auto getA = [&](int kt) -> v16bf {
        if (kt < 8) return load_afrag_lds(comb0h, SH0, mrow, kt * 32 + hi8);
        return (kt == 8) ? xf0 : xf1;
      };
      v16bf af = getA(0);
#pragma unroll
      for (int kt = 0; kt < W0KT; ++kt) {
        v16bf afn = af;
        if (kt + 1 < W0KT) afn = getA(kt + 1);   // software pipeline: hide LDS latency
#pragma unroll
        for (int nt = 0; nt < 4; ++nt) {
          v16bf bf = load_bfrag(wz0p, (w * 4 + nt) * W0KT + kt, lane);
          gacc[nt] = wmma_bf16(af, bf, gacc[nt]);
        }
        af = afn;
      }
    }
    __syncthreads();

    // waves 0-3: park z ; waves 4-7: rc = r * h0
#pragma unroll
    for (int nt = 0; nt < 4; ++nt) {
      int c = w * 64 + nt * 16 + mrow;
#pragma unroll
      for (int j = 0; j < 8; ++j) {
        int row  = j + hi8;
        float gv = sigmoid_f(gacc[nt][j]);
        if (w < 4) {
          zbuf[row * HID + c] = gv;
        } else {
          int hc = c - 256;
          float hold = bf2f(comb0h[row * SH0 + hc]);
          rc[row * SH0 + hc] = f2bf(gv * hold);
        }
      }
    }
    __syncthreads();

    // ---- layer0 candidate: cols w*32..w*32+31 ----
    v8f cacc[2];
#pragma unroll
    for (int nt = 0; nt < 2; ++nt) cacc[nt] = splat8(bcv0[nt]);
    {
      auto getA = [&](int kt) -> v16bf {
        if (kt < 8) return load_afrag_lds(rc, SH0, mrow, kt * 32 + hi8);
        return (kt == 8) ? xf0 : xf1;
      };
      v16bf af = getA(0);
#pragma unroll
      for (int kt = 0; kt < W0KT; ++kt) {
        v16bf afn = af;
        if (kt + 1 < W0KT) afn = getA(kt + 1);
#pragma unroll
        for (int nt = 0; nt < 2; ++nt) {
          v16bf bf = load_bfrag(wc0p, (w * 2 + nt) * W0KT + kt, lane);
          cacc[nt] = wmma_bf16(af, bf, cacc[nt]);
        }
        af = afn;
      }
    }

    // blend h0 = (1-z)h0 + z*tanh(cand); write into comb0h and comb1 second half
#pragma unroll
    for (int nt = 0; nt < 2; ++nt) {
      int c = w * 32 + nt * 16 + mrow;
#pragma unroll
      for (int j = 0; j < 8; ++j) {
        int row = j + hi8;
        float ht   = tanh_f(cacc[nt][j]);
        float z    = zbuf[row * HID + c];
        float hold = bf2f(comb0h[row * SH0 + c]);
        unsigned short hb = f2bf((1.f - z) * hold + z * ht);
        comb0h[row * SH0 + c] = hb;
        comb1[row * SH1 + 256 + c] = hb;
      }
    }
    __syncthreads();

    // ---- layer1 gates: K=512 over comb1 = [h1 | h0_new] ----
#pragma unroll
    for (int nt = 0; nt < 4; ++nt) gacc[nt] = splat8(bzv1[nt]);
    {
      auto getA = [&](int kt) -> v16bf {
        return load_afrag_lds(comb1, SH1, mrow, kt * 32 + hi8);
      };
      v16bf af = getA(0);
#pragma unroll
      for (int kt = 0; kt < W1KT; ++kt) {
        v16bf afn = af;
        if (kt + 1 < W1KT) afn = getA(kt + 1);
#pragma unroll
        for (int nt = 0; nt < 4; ++nt) {
          v16bf bf = load_bfrag(wz1p, (w * 4 + nt) * W1KT + kt, lane);
          gacc[nt] = wmma_bf16(af, bf, gacc[nt]);
        }
        af = afn;
      }
    }
    __syncthreads();

#pragma unroll
    for (int nt = 0; nt < 4; ++nt) {
      int c = w * 64 + nt * 16 + mrow;
#pragma unroll
      for (int j = 0; j < 8; ++j) {
        int row  = j + hi8;
        float gv = sigmoid_f(gacc[nt][j]);
        if (w < 4) {
          zbuf[row * HID + c] = gv;
        } else {
          int hc = c - 256;
          float hold = bf2f(comb1[row * SH1 + hc]);   // h1 old
          rc[row * SH0 + hc] = f2bf(gv * hold);
        }
      }
    }
    __syncthreads();

    // ---- layer1 candidate: K=512 over [r*h1 | h0_new] ----
#pragma unroll
    for (int nt = 0; nt < 2; ++nt) cacc[nt] = splat8(bcv1[nt]);
    {
      auto getA = [&](int kt) -> v16bf {
        if (kt < 8) return load_afrag_lds(rc, SH0, mrow, kt * 32 + hi8);
        return load_afrag_lds(comb1, SH1, mrow, kt * 32 + hi8);
      };
      v16bf af = getA(0);
#pragma unroll
      for (int kt = 0; kt < W1KT; ++kt) {
        v16bf afn = af;
        if (kt + 1 < W1KT) afn = getA(kt + 1);
#pragma unroll
        for (int nt = 0; nt < 2; ++nt) {
          v16bf bf = load_bfrag(wc1p, (w * 2 + nt) * W1KT + kt, lane);
          cacc[nt] = wmma_bf16(af, bf, cacc[nt]);
        }
        af = afn;
      }
    }

    // blend h1, write into comb1 first half
#pragma unroll
    for (int nt = 0; nt < 2; ++nt) {
      int c = w * 32 + nt * 16 + mrow;
#pragma unroll
      for (int j = 0; j < 8; ++j) {
        int row = j + hi8;
        float ht   = tanh_f(cacc[nt][j]);
        float z    = zbuf[row * HID + c];
        float hold = bf2f(comb1[row * SH1 + c]);
        comb1[row * SH1 + c] = f2bf((1.f - z) * hold + z * ht);
      }
    }
    WAIT_ASYNC();          // retire next-x prefetch before the barrier
    __syncthreads();
  }

  // ---- LayerNorm over final h1 (wave w handles rows 2w, 2w+1) ----
#pragma unroll
  for (int rr = 0; rr < 2; ++rr) {
    int row = w * 2 + rr;
    float s = 0.f, s2 = 0.f;
#pragma unroll
    for (int j = 0; j < 8; ++j) {
      float v = bf2f(comb1[row * SH1 + lane + j * 32]);
      s += v; s2 += v * v;
    }
#pragma unroll
    for (int off = 16; off > 0; off >>= 1) {
      s  += __shfl_xor(s,  off, 32);
      s2 += __shfl_xor(s2, off, 32);
    }
    float mean = s * (1.f / 256.f);
    float var  = s2 * (1.f / 256.f) - mean * mean;
    float inv  = rsqrtf(var + LN_EPS);
#pragma unroll
    for (int j = 0; j < 8; ++j) {
      int c = lane + j * 32;
      float v = bf2f(comb1[row * SH1 + c]);
      out[(size_t)(g * 16 + row) * HID + c] = (v - mean) * inv * gamma[c] + beta[c];
    }
  }
}

extern "C" void kernel_launch(void* const* d_in, const int* in_sizes, int n_in,
                              void* d_out, int out_size, void* d_ws, size_t ws_size,
                              hipStream_t stream) {
  (void)in_sizes; (void)n_in; (void)out_size; (void)ws_size;
  const float* x     = (const float*)d_in[0];
  const float* Wz0   = (const float*)d_in[1];
  const float* bz0   = (const float*)d_in[2];
  const float* Wc0   = (const float*)d_in[3];
  const float* bc0   = (const float*)d_in[4];
  const float* Wz1   = (const float*)d_in[5];
  const float* bz1   = (const float*)d_in[6];
  const float* Wc1   = (const float*)d_in[7];
  const float* bc1   = (const float*)d_in[8];
  const float* gamma = (const float*)d_in[9];
  const float* beta  = (const float*)d_in[10];

  unsigned short* ws   = (unsigned short*)d_ws;
  unsigned short* wz0p = ws;                         // 512 x 320 halves
  unsigned short* wc0p = wz0p + (size_t)512 * 320;   // 256 x 320
  unsigned short* wz1p = wc0p + (size_t)256 * 320;   // 512 x 512
  unsigned short* wc1p = wz1p + (size_t)512 * 512;   // 256 x 512

  auto pack = [&](const float* W, int fanout, int Kt, int Nt, unsigned short* dst) {
    int threads = Kt * Nt * 32;
    pack_weights<<<(threads + 255) / 256, 256, 0, stream>>>(W, fanout, Kt, Nt, dst);
  };
  pack(Wz0, 768, W0KT, 32, wz0p);   // z|r columns only (first 512 of 768)
  pack(Wc0, 256, W0KT, 16, wc0p);
  pack(Wz1, 768, W1KT, 32, wz1p);
  pack(Wc1, 256, W1KT, 16, wc1p);

  gru_persistent<<<16, 256, 0, stream>>>(x, bz0, bc0, bz1, bc1,
                                         wz0p, wc0p, wz1p, wc1p,
                                         gamma, beta, (float*)d_out);
}
